// DiffuCoderAttention_70385924046925
// MI455X (gfx1250) — matrix operand order repaired and checked by reading, not compiled, not run on hardware
//
#include <hip/hip_runtime.h>
#include <stdint.h>

// ---------------------------------------------------------------------------
// Types for CDNA5 WMMA (gfx1250, wave32)
// ---------------------------------------------------------------------------
typedef __attribute__((ext_vector_type(16))) __bf16 v16bf;
typedef __attribute__((ext_vector_type(8)))  float  v8f;

union Frag16 {
    v16bf v;
    uint4 q[2];
    unsigned short h[16];
};

__device__ __forceinline__ unsigned short f2bf(float x) {
    union { float f; unsigned u; } c; c.f = x;
    unsigned u = c.u;
    u += 0x7FFFu + ((u >> 16) & 1u);   // round-to-nearest-even
    return (unsigned short)(u >> 16);
}

__device__ __forceinline__ float rmax16(float v) {
    v = fmaxf(v, __shfl_xor(v, 1, 32));
    v = fmaxf(v, __shfl_xor(v, 2, 32));
    v = fmaxf(v, __shfl_xor(v, 4, 32));
    v = fmaxf(v, __shfl_xor(v, 8, 32));
    return v;
}
__device__ __forceinline__ float rsum16(float v) {
    v += __shfl_xor(v, 1, 32);
    v += __shfl_xor(v, 2, 32);
    v += __shfl_xor(v, 4, 32);
    v += __shfl_xor(v, 8, 32);
    return v;
}

// ---------------------------------------------------------------------------
// CDNA5 async copy: GLOBAL_LOAD_ASYNC_TO_LDS_B128 (ASYNCcnt-tracked).
// lds_off = LDS byte address (low 32 bits of the generic pointer, per ISA
// flat->LDS mapping), gptr = 64-bit global address in a VGPR pair.
// ---------------------------------------------------------------------------
__device__ __forceinline__ void async_load_b128(unsigned lds_off, const void* gptr) {
    asm volatile("global_load_async_to_lds_b128 %0, %1, off"
                 :: "v"(lds_off), "v"(gptr) : "memory");
}
__device__ __forceinline__ void wait_async0() {
    asm volatile("s_wait_asynccnt 0x0" ::: "memory");
}
__device__ __forceinline__ unsigned lds_addr(const void* p) {
    return (unsigned)(size_t)p;
}

// ---------------------------------------------------------------------------
// fp32 -> bf16 conversion (vectorized x4)
// ---------------------------------------------------------------------------
__global__ void cvt_f32_bf16(const float* __restrict__ src,
                             unsigned short* __restrict__ dst, int n4) {
    int i = blockIdx.x * blockDim.x + threadIdx.x;
    if (i >= n4) return;
    float4 f = ((const float4*)src)[i];
    ushort4 o;
    o.x = f2bf(f.x); o.y = f2bf(f.y); o.z = f2bf(f.z); o.w = f2bf(f.w);
    ((ushort4*)dst)[i] = o;
}

// ---------------------------------------------------------------------------
// bf16 WMMA GEMM: C[M][N] f32 = A[M][K] bf16 x B[K][N] bf16
// Block tile 256x128, 256 threads (8 waves), wave tile 64x64 (4x4 WMMA tiles):
// 16 WMMAs per 32-K step vs 16 ds_load_b128 -> 2x the FLOP/LDS-byte of a
// 32x32 wave tile. A tile staged with async-to-LDS; B staged transposed
// ([n][k], stride 40 elements = 80B, 16B aligned) via ds_store_b16.
// ---------------------------------------------------------------------------
#define BM 256
#define BN 128
#define BK 32
#define LDS_S 40

__global__ __launch_bounds__(256) void gemm_bf16(
        const unsigned short* __restrict__ A,
        const unsigned short* __restrict__ B,
        float* __restrict__ C, int M, int N, int K) {
    __shared__ __align__(16) unsigned short As[BM * LDS_S];   // 20.0 KB
    __shared__ __align__(16) unsigned short Bs[BN * LDS_S];   // 10.0 KB

    const int tid  = threadIdx.x;
    const int lane = tid & 31;
    const int wave = tid >> 5;
    const int waveRow = wave >> 1;      // 0..3 -> 64-row slices
    const int waveCol = wave & 1;       // 0..1 -> 64-col slices
    const int rl   = lane & 15;
    const int half = lane >> 4;

    const int bm = blockIdx.x * BM;
    const int bn = blockIdx.y * BN;

    v8f acc[4][4];
#pragma unroll
    for (int r = 0; r < 4; r++)
#pragma unroll
        for (int c = 0; c < 4; c++)
#pragma unroll
            for (int j = 0; j < 8; j++) acc[r][c][j] = 0.0f;

    // loader mapping: A -> one full 32-wide k row per thread (4 x b128 async)
    //                 B -> 16 n-elements of one k row, stored transposed
    const int arow = tid;               // 0..255
    const int bk_  = tid >> 3;          // 0..31
    const int bnb  = (tid & 7) * 16;    // 0,16,..,112

    const unsigned short* aG = A + (size_t)(bm + arow) * K;
    const unsigned aL = lds_addr(&As[arow * LDS_S]);

    for (int k0 = 0; k0 < K; k0 += BK) {
        // --- stage A tile: 4 async b128 per thread ---
#pragma unroll
        for (int i = 0; i < 4; i++)
            async_load_b128(aL + i * 16, aG + k0 + i * 8);

        // --- stage B tile transposed: Bs[n][k] ---
        const unsigned short* bg = B + (size_t)(k0 + bk_) * N + bn + bnb;
        unsigned short tmp[16];
        *(uint4*)(tmp)     = ((const uint4*)bg)[0];
        *(uint4*)(tmp + 8) = ((const uint4*)bg)[1];
#pragma unroll
        for (int i = 0; i < 16; i++) Bs[(bnb + i) * LDS_S + bk_] = tmp[i];

        if (k0 + BK < K) {
            __builtin_prefetch(aG + k0 + BK, 0, 0);
            __builtin_prefetch(B + (size_t)(k0 + BK + bk_) * N + bn + bnb, 0, 0);
        }
        wait_async0();
        __syncthreads();

        // --- preload the 4 B fragments for this wave's 64 columns ---
        Frag16 bfr[4];
#pragma unroll
        for (int ct = 0; ct < 4; ct++) {
            const int n = waveCol * 64 + ct * 16 + rl;
            bfr[ct].q[0] = *(const uint4*)(&Bs[n * LDS_S + half * 8]);
            bfr[ct].q[1] = *(const uint4*)(&Bs[n * LDS_S + 16 + half * 8]);
        }
#pragma unroll
        for (int rt = 0; rt < 4; rt++) {
            Frag16 a;
            const int row = waveRow * 64 + rt * 16 + rl;
            a.q[0] = *(const uint4*)(&As[row * LDS_S + half * 8]);
            a.q[1] = *(const uint4*)(&As[row * LDS_S + 16 + half * 8]);
#pragma unroll
            for (int ct = 0; ct < 4; ct++) {
                acc[rt][ct] = __builtin_amdgcn_wmma_f32_16x16x32_bf16(
                    false, a.v, false, bfr[ct].v, (short)0, acc[rt][ct], false, false);
            }
        }
        __syncthreads();
    }

#pragma unroll
    for (int rt = 0; rt < 4; rt++)
#pragma unroll
        for (int ct = 0; ct < 4; ct++)
#pragma unroll
            for (int j = 0; j < 8; j++) {
                const int row = bm + waveRow * 64 + rt * 16 + half * 8 + j;
                const int col = bn + waveCol * 64 + ct * 16 + rl;
                C[(size_t)row * N + col] = acc[rt][ct][j];
            }
}

// ---------------------------------------------------------------------------
// RoPE (interleaved, first half of head_dim) + scale + relayout to
// [head][s][128] bf16.  X is f32 [S][nH*128] (ld = nH*128).
// ---------------------------------------------------------------------------
__global__ void rope_pack(const float* __restrict__ X,
                          unsigned short* __restrict__ Out,
                          int S, int nH, int ld, float scale, int do_rope) {
    int idx = blockIdx.x * blockDim.x + threadIdx.x;   // over S*nH*128
    int d  = idx & 127;
    int t  = idx >> 7;
    int hh = t % nH;
    int s  = t / nH;
    if (s >= S) return;
    const float* row = X + (size_t)s * ld + hh * 128;
    float val;
    if (do_rope && d < 64) {
        int p = d >> 1;
        float x0 = row[2 * p], x1 = row[2 * p + 1];
        float inv = __powf(10000.0f, -(float)(2 * p) / 64.0f);
        float ang = (float)s * inv;
        float c = __cosf(ang), sn = __sinf(ang);
        val = (d & 1) ? (x0 * sn + x1 * c) : (x0 * c - x1 * sn);
    } else {
        val = row[d];
    }
    Out[(size_t)hh * S * 128 + (size_t)s * 128 + d] = f2bf(val * scale);
}

// ---------------------------------------------------------------------------
// Flash attention, one head x 128 queries per block (8 waves x 16 rows).
// K/V tiles of 32 keys staged in LDS (K via async-to-LDS, V transposed via
// ds_store_b16), shared by all 8 waves. Online softmax per 16-row tile.
// Scale 1/sqrt(128) pre-folded into Q. Output bf16 flat [s][h*128+d].
// ---------------------------------------------------------------------------
#define HD 128
#define KT_S 136   // Kt stride in elements (272B, 16B aligned)
#define VT_S 40    // Vt stride (keys dim padded 32->40)
#define PT_S 40

__global__ __launch_bounds__(256) void attn_kernel(
        const unsigned short* __restrict__ Qb,
        const unsigned short* __restrict__ Kb,
        const unsigned short* __restrict__ Vb,
        unsigned short* __restrict__ Oflat, int S) {
    __shared__ __align__(16) unsigned short Kt[32 * KT_S];     // [key][d]  8.5 KB
    __shared__ __align__(16) unsigned short Vt[HD * VT_S];     // [d][key] 10.0 KB
    __shared__ __align__(16) unsigned short Pt[8][16 * PT_S];  // per-wave P 10.0 KB

    const int h   = blockIdx.x;     // 0..15
    const int qt  = blockIdx.y;     // 0..S/128-1
    const int kvh = h >> 2;         // GQA: 4 query heads per kv head
    const int tid  = threadIdx.x;
    const int lane = tid & 31;
    const int wave = tid >> 5;
    const int rl   = lane & 15;
    const int half = lane >> 4;

    const unsigned short* Qh = Qb + (size_t)h   * S * HD;
    const unsigned short* Kh = Kb + (size_t)kvh * S * HD;
    const unsigned short* Vh = Vb + (size_t)kvh * S * HD;

    const int qbase = qt * 128 + wave * 16;

    // Q A-fragments: 4 chunks of K=32 over head_dim
    Frag16 aq[4];
#pragma unroll
    for (int c = 0; c < 4; c++) {
        const unsigned short* p = Qh + (size_t)(qbase + rl) * HD + c * 32 + half * 8;
        aq[c].q[0] = *(const uint4*)p;
        aq[c].q[1] = *(const uint4*)(p + 16);
    }

    float mrow[8], lrow[8];
    v8f Oacc[8];
#pragma unroll
    for (int j = 0; j < 8; j++) { mrow[j] = -1e30f; lrow[j] = 0.0f; }
#pragma unroll
    for (int n = 0; n < 8; n++)
#pragma unroll
        for (int j = 0; j < 8; j++) Oacc[n][j] = 0.0f;

    // loader mapping (256 threads, 32 keys x 128 d = 16 elems/thread)
    const int lkey = tid >> 3;          // 0..31
    const int ldb  = (tid & 7) * 16;    // 0,16,..,112
    const unsigned kL = lds_addr(&Kt[lkey * KT_S + ldb]);

    for (int kb = 0; kb < S; kb += 32) {
        __syncthreads();   // previous tiles fully consumed
        {   // stage K tile [key][d] via async-to-LDS (2 x b128 per thread)
            const unsigned short* ks = Kh + (size_t)(kb + lkey) * HD + ldb;
            async_load_b128(kL,      ks);
            async_load_b128(kL + 16, ks + 8);
            // stage V tile transposed [d][key]
            unsigned short vtmp[16];
            const uint4* vs = (const uint4*)(Vh + (size_t)(kb + lkey) * HD + ldb);
            *(uint4*)(vtmp)     = vs[0];
            *(uint4*)(vtmp + 8) = vs[1];
#pragma unroll
            for (int i = 0; i < 16; i++) Vt[(ldb + i) * VT_S + lkey] = vtmp[i];
        }
        if (kb + 32 < S) {
            __builtin_prefetch(Kh + (size_t)(kb + 32 + lkey) * HD + ldb, 0, 0);
            __builtin_prefetch(Vh + (size_t)(kb + 32 + lkey) * HD + ldb, 0, 0);
        }
        wait_async0();
        __syncthreads();

        // ---- scores: S0 (keys kb..kb+15), S1 (keys kb+16..kb+31) ----
        v8f s0, s1;
#pragma unroll
        for (int j = 0; j < 8; j++) { s0[j] = 0.0f; s1[j] = 0.0f; }
#pragma unroll
        for (int c = 0; c < 4; c++) {
            Frag16 bk0, bk1;
            bk0.q[0] = *(const uint4*)(&Kt[rl * KT_S + c * 32 + half * 8]);
            bk0.q[1] = *(const uint4*)(&Kt[rl * KT_S + c * 32 + 16 + half * 8]);
            s0 = __builtin_amdgcn_wmma_f32_16x16x32_bf16(
                false, aq[c].v, false, bk0.v, (short)0, s0, false, false);
            bk1.q[0] = *(const uint4*)(&Kt[(16 + rl) * KT_S + c * 32 + half * 8]);
            bk1.q[1] = *(const uint4*)(&Kt[(16 + rl) * KT_S + c * 32 + 16 + half * 8]);
            s1 = __builtin_amdgcn_wmma_f32_16x16x32_bf16(
                false, aq[c].v, false, bk1.v, (short)0, s1, false, false);
        }

        // ---- online softmax update (rows live across 16-lane halves) ----
        float p0[8], p1[8];
#pragma unroll
        for (int j = 0; j < 8; j++) {
            float rm = rmax16(fmaxf(s0[j], s1[j]));
            float mn = fmaxf(mrow[j], rm);
            float alpha = __expf(mrow[j] - mn);
            mrow[j] = mn;
            p0[j] = __expf(s0[j] - mn);
            p1[j] = __expf(s1[j] - mn);
            float rs = rsum16(p0[j] + p1[j]);
            lrow[j] = lrow[j] * alpha + rs;
#pragma unroll
            for (int n = 0; n < 8; n++) Oacc[n][j] *= alpha;
        }

        // ---- P (C layout f32) -> LDS -> A layout bf16 ----
#pragma unroll
        for (int j = 0; j < 8; j++) {
            const int prow = half * 8 + j;
            Pt[wave][prow * PT_S + rl]      = f2bf(p0[j]);
            Pt[wave][prow * PT_S + 16 + rl] = f2bf(p1[j]);
        }
        Frag16 ap;
        ap.q[0] = *(const uint4*)(&Pt[wave][rl * PT_S + half * 8]);
        ap.q[1] = *(const uint4*)(&Pt[wave][rl * PT_S + 16 + half * 8]);

        // ---- O += P @ V ----
#pragma unroll
        for (int n = 0; n < 8; n++) {
            Frag16 bv;
            bv.q[0] = *(const uint4*)(&Vt[(n * 16 + rl) * VT_S + half * 8]);
            bv.q[1] = *(const uint4*)(&Vt[(n * 16 + rl) * VT_S + 16 + half * 8]);
            Oacc[n] = __builtin_amdgcn_wmma_f32_16x16x32_bf16(
                false, ap.v, false, bv.v, (short)0, Oacc[n], false, false);
        }
    }

    // ---- epilogue: normalize, write bf16 flat [s][h*128+d] ----
#pragma unroll
    for (int j = 0; j < 8; j++) {
        const float inv_l = 1.0f / lrow[j];
        const int srow = qbase + half * 8 + j;
#pragma unroll
        for (int n = 0; n < 8; n++) {
            const int col = h * HD + n * 16 + rl;
            Oflat[(size_t)srow * 2048 + col] = f2bf(Oacc[n][j] * inv_l);
        }
    }
}

// ---------------------------------------------------------------------------
// Host-side launch sequence (all on `stream`, graph-capture safe)
// ---------------------------------------------------------------------------
extern "C" void kernel_launch(void* const* d_in, const int* in_sizes, int n_in,
                              void* d_out, int out_size, void* d_ws, size_t ws_size,
                              hipStream_t stream) {
    const float* hs = (const float*)d_in[0];   // [S][2048]
    const float* Wq = (const float*)d_in[1];   // [2048][2048]
    const float* Wk = (const float*)d_in[2];   // [2048][512]
    const float* Wv = (const float*)d_in[3];   // [2048][512]
    const float* Wo = (const float*)d_in[4];   // [2048][2048]
    float* out = (float*)d_out;

    const int S = 2048, H = 2048, NH = 16, NKV = 4, HDIM = 128;
    const int NKVD = NKV * HDIM;   // 512

    char* p = (char*)d_ws;
    auto take = [&](size_t bytes) {
        char* r = p; p += (bytes + 255) & ~(size_t)255; return r;
    };
    unsigned short* Xb  = (unsigned short*)take((size_t)S * H * 2);
    unsigned short* Wqb = (unsigned short*)take((size_t)H * H * 2);
    unsigned short* Wkb = (unsigned short*)take((size_t)H * NKVD * 2);
    unsigned short* Wvb = (unsigned short*)take((size_t)H * NKVD * 2);
    unsigned short* Wob = (unsigned short*)take((size_t)H * H * 2);
    float*          Qf  = (float*)take((size_t)S * H * 4);
    float*          Kf  = (float*)take((size_t)S * NKVD * 4);
    float*          Vf  = (float*)take((size_t)S * NKVD * 4);
    unsigned short* Qb  = (unsigned short*)take((size_t)NH  * S * HDIM * 2);
    unsigned short* Kb  = (unsigned short*)take((size_t)NKV * S * HDIM * 2);
    unsigned short* Vb  = (unsigned short*)take((size_t)NKV * S * HDIM * 2);
    unsigned short* Af  = (unsigned short*)take((size_t)S * H * 2);

    // 1) fp32 -> bf16
    cvt_f32_bf16<<<(S * H / 4 + 255) / 256, 256, 0, stream>>>(hs, Xb, S * H / 4);
    cvt_f32_bf16<<<(H * H / 4 + 255) / 256, 256, 0, stream>>>(Wq, Wqb, H * H / 4);
    cvt_f32_bf16<<<(H * NKVD / 4 + 255) / 256, 256, 0, stream>>>(Wk, Wkb, H * NKVD / 4);
    cvt_f32_bf16<<<(H * NKVD / 4 + 255) / 256, 256, 0, stream>>>(Wv, Wvb, H * NKVD / 4);
    cvt_f32_bf16<<<(H * H / 4 + 255) / 256, 256, 0, stream>>>(Wo, Wob, H * H / 4);

    // 2) QKV projections (bf16 WMMA, f32 out)
    gemm_bf16<<<dim3(S / BM, H / BN),    256, 0, stream>>>(Xb, Wqb, Qf, S, H,    H);
    gemm_bf16<<<dim3(S / BM, NKVD / BN), 256, 0, stream>>>(Xb, Wkb, Kf, S, NKVD, H);
    gemm_bf16<<<dim3(S / BM, NKVD / BN), 256, 0, stream>>>(Xb, Wvb, Vf, S, NKVD, H);

    // 3) RoPE + scale + relayout to [head][s][128] bf16
    const float qscale = 0.08838834764831845f;  // 1/sqrt(128)
    rope_pack<<<(S * NH  * HDIM) / 256, 256, 0, stream>>>(Qf, Qb, S, NH,  H,    qscale, 1);
    rope_pack<<<(S * NKV * HDIM) / 256, 256, 0, stream>>>(Kf, Kb, S, NKV, NKVD, 1.0f,   1);
    rope_pack<<<(S * NKV * HDIM) / 256, 256, 0, stream>>>(Vf, Vb, S, NKV, NKVD, 1.0f,   0);

    // 4) flash attention -> bf16 flat [s][2048]
    attn_kernel<<<dim3(NH, S / 128), 256, 0, stream>>>(Qb, Kb, Vb, Af, S);

    // 5) output projection -> f32 d_out
    gemm_bf16<<<dim3(S / BM, H / BN), 256, 0, stream>>>(Af, Wob, out, S, H, H);
}
